// RNNBlock_1632087572931
// MI455X (gfx1250) — matrix operand order repaired and checked by reading, not compiled, not run on hardware
//
#include <hip/hip_runtime.h>

// ---------------------------------------------------------------------------
// RNN block for MI455X (gfx1250, wave32, WMMA).
//   T=512, B=64, D=1024, H=1024
//   prep : W_xh, W_hh -> transposed bf16 copies in ws (L2-resident),
//          zero bf16 h0 ping-pong buffer
//   proj : xproj = input @ W_xh + b_h   (WMMA bf16, written into d_out slabs)
//          A staged f32->bf16 through double-buffered LDS, one barrier/K-step
//   recur: h_t = tanh(xproj_t + h_{t-1} @ W_hh), 512 sequential launches.
//          LDS-free inner loop: 4x global_load_b128 + 1x wmma per K-step.
//   state: d2d async copy of last slab.
// ---------------------------------------------------------------------------

typedef __attribute__((ext_vector_type(16))) __bf16 v16bf;
typedef __attribute__((ext_vector_type(8)))  float  v8f;
typedef unsigned int u32x4 __attribute__((ext_vector_type(4)));
typedef float        f32x4 __attribute__((ext_vector_type(4)));

union Frag {
    v16bf v;
    u32x4 q[2];   // two 16-byte halves; matches element order of the layouts
};

static __device__ __forceinline__ unsigned short f32_to_bf16(float f) {
    unsigned int u = __float_as_uint(f);
    u = (u + 0x7FFFu + ((u >> 16) & 1u)) >> 16;   // round-to-nearest-even
    return (unsigned short)u;
}

static __device__ __forceinline__ unsigned int pack_bf16x2(float lo, float hi) {
    return (unsigned int)f32_to_bf16(lo) | ((unsigned int)f32_to_bf16(hi) << 16);
}

// ---- prep: WT[n*1024 + k] = bf16(W[k*1024 + n])  (K == N == 1024) ----------
__global__ void __launch_bounds__(256)
transpose_to_bf16(const float* __restrict__ W, unsigned short* __restrict__ WT) {
    const int i = blockIdx.x * 256 + threadIdx.x;   // over 1024*1024
    const int k = i >> 10;
    const int n = i & 1023;
    WT[(size_t)n * 1024 + k] = f32_to_bf16(W[i]);
}

__global__ void __launch_bounds__(256)
zero_bf16(unsigned short* __restrict__ p) {
    p[blockIdx.x * 256 + threadIdx.x] = 0;
}

// ---- projection: out[m,n] = sum_k Xin[m,k]*W[k,n] + bias[n] ----------------
// Block: 256 threads = 8 waves. Tile: 32 (M) x 128 (N); wave w owns columns
// [n0+16w, n0+16w+16) and two M-subtiles sharing one B fragment.
__global__ void __launch_bounds__(256)
proj_wmma(const float* __restrict__ Xin,            // [M, K] f32
          const unsigned short* __restrict__ WT,    // [N, K] bf16 (transposed)
          const float* __restrict__ bias,           // [N]
          float* __restrict__ out,                  // [M, N]
          int M, int K, int N) {
    // double-buffered A tile: 32 M-rows x 32 K bf16, rows padded to 40
    __shared__ unsigned short As[2][32 * 40];

    const int tid  = threadIdx.x;
    const int wave = tid >> 5;
    const int lane = tid & 31;
    const int half = lane >> 4;
    const int l16  = lane & 15;
    const int n0 = blockIdx.x * 128;
    const int m0 = blockIdx.y * 32;

    // staging slot: thread owns row sr, cols [sc, sc+4) of the 32x32 tile
    const int sr = tid >> 3;          // 0..31
    const int sc = (tid & 7) * 4;     // 0,4,..,28
    const float* xrow = Xin + (size_t)(m0 + sr) * K + sc;   // + k0 each step
    unsigned int* sdst[2] = { (unsigned int*)&As[0][sr * 40 + sc],
                              (unsigned int*)&As[1][sr * 40 + sc] };

    // fragment read pointers (fixed; content changes per K-step)
    const unsigned short* a0p[2] = { &As[0][l16 * 40 + half * 8],
                                     &As[1][l16 * 40 + half * 8] };
    const unsigned short* a1p[2] = { &As[0][(16 + l16) * 40 + half * 8],
                                     &As[1][(16 + l16) * 40 + half * 8] };

    const int ncol = n0 + wave * 16 + l16;
    const unsigned short* wrow = WT + (size_t)ncol * K + half * 8;

    v8f acc0 = {}, acc1 = {};

    for (int k0 = 0; k0 < K; k0 += 32) {
        const int p = (k0 >> 5) & 1;

        // stage A: one b128 load, pack to bf16, two ds_store_b32
        const f32x4 xv = *(const f32x4*)(xrow + k0);
        sdst[p][0] = pack_bf16x2(xv.x, xv.y);
        sdst[p][1] = pack_bf16x2(xv.z, xv.w);

        __syncthreads();   // stores(p) visible; prior-step reads (1-p) done

        // B fragment: column ncol, K = half*8 .. +7 and 16+half*8 .. +7 in WT row.
        Frag fb;
        fb.q[0] = *(const u32x4*)(wrow + k0);
        fb.q[1] = *(const u32x4*)(wrow + k0 + 16);

        // A fragments: rows l16 / l16+16, two contiguous 8-element runs.
        Frag fa0, fa1;
        fa0.q[0] = *(const u32x4*)(a0p[p]);
        fa0.q[1] = *(const u32x4*)(a0p[p] + 16);
        fa1.q[0] = *(const u32x4*)(a1p[p]);
        fa1.q[1] = *(const u32x4*)(a1p[p] + 16);

        acc0 = __builtin_amdgcn_wmma_f32_16x16x32_bf16(
            false, fa0.v, false, fb.v, (short)0, acc0, false, false);
        acc1 = __builtin_amdgcn_wmma_f32_16x16x32_bf16(
            false, fa1.v, false, fb.v, (short)0, acc1, false, false);
    }

    const float bv = bias[ncol];
#pragma unroll
    for (int r = 0; r < 8; ++r) {
        const int m = m0 + half * 8 + r;         // C/D layout: M = half*8 + r
        out[(size_t)m * N + ncol]        = acc0[r] + bv;
        out[(size_t)(m + 16) * N + ncol] = acc1[r] + bv;
    }
}

// ---- recurrence step: slab = tanh(slab + Hprev @ W_hh); Hnext = bf16(slab) -
// Grid: 8 (N/128) x 4 (B/16), 256 threads. LDS-free inner loop.
__global__ void __launch_bounds__(256)
recur_wmma(const unsigned short* __restrict__ Hprev,  // [B, K] bf16
           const unsigned short* __restrict__ WT,     // [N, K] bf16 (transposed)
           float* __restrict__ slab,                  // [B, N] f32: xproj in, h out
           unsigned short* __restrict__ Hnext,        // [B, N] bf16 out
           int K, int N) {
    const int tid  = threadIdx.x;
    const int wave = tid >> 5;
    const int lane = tid & 31;
    const int half = lane >> 4;
    const int l16  = lane & 15;
    const int n0 = blockIdx.x * 128;
    const int m0 = blockIdx.y * 16;

    const int ncol = n0 + wave * 16 + l16;
    const unsigned short* wrow = WT + (size_t)ncol * K + half * 8;
    const unsigned short* arow = Hprev + (size_t)(m0 + l16) * K + half * 8;

    v8f acc = {};
#pragma unroll 4
    for (int k0 = 0; k0 < K; k0 += 32) {
        Frag fa, fb;
        fa.q[0] = *(const u32x4*)(arow + k0);
        fa.q[1] = *(const u32x4*)(arow + k0 + 16);
        fb.q[0] = *(const u32x4*)(wrow + k0);
        fb.q[1] = *(const u32x4*)(wrow + k0 + 16);
        acc = __builtin_amdgcn_wmma_f32_16x16x32_bf16(
            false, fa.v, false, fb.v, (short)0, acc, false, false);
    }

#pragma unroll
    for (int r = 0; r < 8; ++r) {
        const int m = m0 + half * 8 + r;
        const size_t idx = (size_t)m * N + ncol;
        const float v = tanhf(acc[r] + slab[idx]);
        slab[idx] = v;
        Hnext[idx] = f32_to_bf16(v);
    }
}

extern "C" void kernel_launch(void* const* d_in, const int* in_sizes, int n_in,
                              void* d_out, int out_size, void* d_ws, size_t ws_size,
                              hipStream_t stream) {
    (void)in_sizes; (void)n_in; (void)out_size; (void)ws_size;

    const int T = 512, B = 64, D = 1024, H = 1024;
    const float* input = (const float*)d_in[0];  // [T,B,D]
    const float* W_xh  = (const float*)d_in[1];  // [D,H]
    const float* W_hh  = (const float*)d_in[2];  // [H,H]
    const float* b_h   = (const float*)d_in[3];  // [H]
    float* out = (float*)d_out;                  // [T,B,H] output then [B,H] state

    // ws layout: WxhT (2MB) | WhhT (2MB) | hb0 (128KB) | hb1 (128KB)
    unsigned short* wxhT = (unsigned short*)d_ws;
    unsigned short* whhT = wxhT + (size_t)D * H;
    unsigned short* hb0  = whhT + (size_t)H * H;
    unsigned short* hb1  = hb0 + (size_t)B * H;

    // 0) prep: transposed bf16 weights + zeroed h_{-1}
    transpose_to_bf16<<<(D * H) / 256, 256, 0, stream>>>(W_xh, wxhT);
    transpose_to_bf16<<<(H * H) / 256, 256, 0, stream>>>(W_hh, whhT);
    zero_bf16<<<(B * H) / 256, 256, 0, stream>>>(hb1);   // prev buffer for t=0

    // 1) xproj = input @ W_xh + b_h  -> straight into output slabs
    {
        const int M = T * B;                     // 32768
        dim3 grid(H / 128, M / 32);              // 8 x 1024
        proj_wmma<<<grid, 256, 0, stream>>>(input, wxhT, b_h, out, M, D, H);
    }

    // 2) 512 sequential recurrence steps (in place over xproj slabs).
    //    bf16 h ping-pong: prev = hb[(t+1)&1], next = hb[t&1].
    for (int t = 0; t < T; ++t) {
        const unsigned short* hprev = ((t + 1) & 1) ? hb1 : hb0;
        unsigned short* hnext       = (t & 1) ? hb1 : hb0;
        float* slab = out + (size_t)t * B * H;
        dim3 grid(H / 128, B / 16);              // 8 x 4 = 32 blocks
        recur_wmma<<<grid, 256, 0, stream>>>(hprev, whhT, slab, hnext, H, H);
    }

    // 3) final state = h_{T-1}
    hipMemcpyAsync(out + (size_t)T * B * H,
                   out + (size_t)(T - 1) * B * H,
                   (size_t)B * H * sizeof(float),
                   hipMemcpyDeviceToDevice, stream);
}
